// MaskedSelfAttention_7069516169386
// MI455X (gfx1250) — matrix operand order repaired and checked by reading, not compiled, not run on hardware
//
#include <hip/hip_runtime.h>
#include <hip/hip_bf16.h>
#include <cstdint>
#include <cstddef>

// ---------------------------------------------------------------------------
// Masked self-attention, MI455X (gfx1250): bf16 WMMA flash-attention pipeline
// with async global->LDS double-buffered staging (ASYNCcnt path).
// ---------------------------------------------------------------------------

typedef __attribute__((ext_vector_type(16))) __bf16 v16bf;
typedef __attribute__((ext_vector_type(8)))  __bf16 v8bf;
typedef __attribute__((ext_vector_type(8)))  float  v8f;
typedef int v4i __attribute__((vector_size(16)));   // matches async-LDS builtin pointee

#define D_EMBED 768
#define D_HEAD  64
#define BATCH   4
#define SEQ     4096
#define BLOCK_M 128                 // rows per workgroup: 8 waves x 16 rows
#define NWAVES  8

// --- workspace layout (bytes) ---
#define WT_ELEMS   (D_HEAD * D_EMBED)          // per matrix, bf16, transposed [64][768]
#define OFF_WT     0
#define QKV_ELEMS  (BATCH * SEQ * D_HEAD)      // bf16
#define OFF_Q      (3 * WT_ELEMS * 2)
#define OFF_K      (OFF_Q + QKV_ELEMS * 2)
#define OFF_VT     (OFF_K + QKV_ELEMS * 2)

// --- CDNA5 async global->LDS copy (16B per lane), with safe fallback ---
#if defined(__has_builtin)
#if __has_builtin(__builtin_amdgcn_global_load_async_to_lds_b128)
#define HAVE_ASYNC_LDS 1
#endif
#endif

static __device__ inline void async_copy16(void* lds_dst, const void* gsrc) {
#if defined(HAVE_ASYNC_LDS)
  __builtin_amdgcn_global_load_async_to_lds_b128(
      (__attribute__((address_space(1))) v4i*)gsrc,
      (__attribute__((address_space(3))) v4i*)lds_dst, 0, 0);
#else
  *(uint4*)lds_dst = *(const uint4*)gsrc;   // synchronous fallback
#endif
}

static __device__ inline void wait_async0() {
#if defined(__has_builtin) && __has_builtin(__builtin_amdgcn_s_wait_asynccnt)
  __builtin_amdgcn_s_wait_asynccnt(0);
#else
  asm volatile("s_wait_asynccnt 0x0" ::: "memory");
#endif
}

static __device__ inline v16bf combine16(v8bf lo, v8bf hi) {
  v16bf r;
#pragma unroll
  for (int i = 0; i < 8; ++i) { r[i] = lo[i]; r[i + 8] = hi[i]; }
  return r;
}

static __device__ inline float hmax16(float x) {
  // butterfly max across a 16-lane half (xor 1,2,4,8 never crosses the half)
  x = fmaxf(x, __shfl_xor(x, 1, 32));
  x = fmaxf(x, __shfl_xor(x, 2, 32));
  x = fmaxf(x, __shfl_xor(x, 4, 32));
  x = fmaxf(x, __shfl_xor(x, 8, 32));
  return x;
}
static __device__ inline float hsum16(float x) {
  x += __shfl_xor(x, 1, 32);
  x += __shfl_xor(x, 2, 32);
  x += __shfl_xor(x, 4, 32);
  x += __shfl_xor(x, 8, 32);
  return x;
}

// ---------------------------------------------------------------------------
// Kernel 0: convert Wq/Wk/Wv fp32 [768][64] -> bf16 transposed [64][768]
// so B-fragment lanes (one N-column each) read two runs of 8 contiguous bf16.
// ---------------------------------------------------------------------------
__global__ void wt_convert(const float* __restrict__ Wq,
                           const float* __restrict__ Wk,
                           const float* __restrict__ Wv,
                           __bf16* __restrict__ wt) {
  int idx = blockIdx.x * blockDim.x + threadIdx.x;
  if (idx >= 3 * WT_ELEMS) return;
  int mat = idx / WT_ELEMS;
  int rem = idx % WT_ELEMS;
  int k   = rem / D_HEAD;
  int n   = rem % D_HEAD;
  const float* W = (mat == 0) ? Wq : (mat == 1) ? Wk : Wv;
  wt[(size_t)mat * WT_ELEMS + (size_t)n * D_EMBED + k] = (__bf16)W[(size_t)k * D_HEAD + n];
}

// ---------------------------------------------------------------------------
// Kernel 1: QKV projection. W panels (shared by all 8 waves) are staged into
// LDS via async copies, double buffered over the 24 k-steps.
// ---------------------------------------------------------------------------
__global__ __launch_bounds__(256) void qkv_proj(
    const float* __restrict__ x, const __bf16* __restrict__ wt,
    const float* __restrict__ bq, const float* __restrict__ bk,
    const float* __restrict__ bv,
    __bf16* __restrict__ Q, __bf16* __restrict__ K, __bf16* __restrict__ Vt) {
  // W panel for one k-chunk: [mat][col][32 k] = 3*64*32 bf16 = 12KB, x2 buffers
  __shared__ __align__(16) __bf16 wpanel[2][3 * D_HEAD * 32];

  const int tid   = threadIdx.x;
  const int lane  = tid & 31;
  const int wid   = tid >> 5;
  const int half  = lane >> 4;          // 0|1
  const int lm    = lane & 15;
  const int b     = blockIdx.x / (SEQ / BLOCK_M);
  const int m0    = (blockIdx.x % (SEQ / BLOCK_M)) * BLOCK_M + wid * 16;
  const int kbase = half * 8;           // ISA 16-bit A-layout K offset per lane half

  v8f acc[3][4];
#pragma unroll
  for (int m = 0; m < 3; ++m)
#pragma unroll
    for (int t = 0; t < 4; ++t) acc[m][t] = v8f{};

  const float* xrow = x + ((size_t)b * SEQ + (m0 + lm)) * D_EMBED;

  // stage a 32-wide k panel of all three W matrices: 768 chunks of 16B,
  // 3 chunks per thread.  chunk ch -> (mat = ch>>8, col = (ch>>2)&63, k8 = ch&3)
  auto stageW = [&](int buf, int kc) {
#pragma unroll
    for (int j = 0; j < 3; ++j) {
      const int ch  = j * 256 + tid;
      const int mat = ch >> 8;
      const int col = (ch >> 2) & 63;
      const int k8  = (ch & 3) * 8;
      async_copy16(&wpanel[buf][(size_t)ch * 8],
                   wt + (size_t)mat * WT_ELEMS + (size_t)col * D_EMBED + kc + k8);
    }
  };

  stageW(0, 0);

  for (int ic = 0; ic < D_EMBED / 32; ++ic) {
    const int kc  = ic * 32;
    const int buf = ic & 1;
    wait_async0();
    __syncthreads();                        // panel(ic) visible; panel(ic-1) consumed
    if (kc + 32 < D_EMBED) stageW(buf ^ 1, kc + 32);

    // A fragment: 16 fp32 -> bf16 (K = kc+kbase..+7 and kc+kbase+16..+23)
    float4 f0 = *(const float4*)(xrow + kc + kbase);
    float4 f1 = *(const float4*)(xrow + kc + kbase + 4);
    float4 f2 = *(const float4*)(xrow + kc + kbase + 16);
    float4 f3 = *(const float4*)(xrow + kc + kbase + 20);
    v16bf a;
    a[0]=(__bf16)f0.x; a[1]=(__bf16)f0.y; a[2]=(__bf16)f0.z; a[3]=(__bf16)f0.w;
    a[4]=(__bf16)f1.x; a[5]=(__bf16)f1.y; a[6]=(__bf16)f1.z; a[7]=(__bf16)f1.w;
    a[8]=(__bf16)f2.x; a[9]=(__bf16)f2.y; a[10]=(__bf16)f2.z; a[11]=(__bf16)f2.w;
    a[12]=(__bf16)f3.x; a[13]=(__bf16)f3.y; a[14]=(__bf16)f3.z; a[15]=(__bf16)f3.w;

#pragma unroll
    for (int m = 0; m < 3; ++m) {
#pragma unroll
      for (int t = 0; t < 4; ++t) {
        // B-layout: lane = column (t*16+lm); K = kc+kbase..+7 and kc+kbase+16..+23
        const __bf16* wp = &wpanel[buf][((m * 64 + t * 16 + lm) * 32) + half * 8];
        v16bf bf = combine16(*(const v8bf*)wp, *(const v8bf*)(wp + 16));
        acc[m][t] = __builtin_amdgcn_wmma_f32_16x16x32_bf16(
            false, a, false, bf, (short)0, acc[m][t], false, false);
      }
    }
  }

  const float* bias[3] = {bq, bk, bv};
#pragma unroll
  for (int m = 0; m < 3; ++m)
#pragma unroll
    for (int t = 0; t < 4; ++t) {
      const int col = t * 16 + lm;
      const float bb = bias[m][col];
#pragma unroll
      for (int v = 0; v < 8; ++v) {
        const int r = m0 + v + 8 * half;       // C-layout: M = v + 8*(lane>=16)
        const float val = acc[m][t][v] + bb;
        if (m == 0)      Q [((size_t)b * SEQ + r) * D_HEAD + col] = (__bf16)val;
        else if (m == 1) K [((size_t)b * SEQ + r) * D_HEAD + col] = (__bf16)val;
        else             Vt[((size_t)b * D_HEAD + col) * SEQ + r] = (__bf16)val;
      }
    }
}

// ---------------------------------------------------------------------------
// Kernel 2: flash attention. K/V tiles (shared by all 8 waves) are staged via
// async global->LDS copies, double buffered. Per 32-key step: 4 score WMMAs,
// mask, online softmax (half-wave butterflies), LDS C->A re-layout, 4 PV WMMAs.
// ---------------------------------------------------------------------------
__global__ __launch_bounds__(256) void attention(
    const __bf16* __restrict__ Q, const __bf16* __restrict__ K,
    const __bf16* __restrict__ Vt, const int* __restrict__ mask,
    float* __restrict__ out) {
  __shared__ __align__(16) __bf16 kbuf[2][32 * D_HEAD];   // [key][dh]   4KB each
  __shared__ __align__(16) __bf16 vbuf[2][D_HEAD * 32];   // [dh][key]   4KB each
  __shared__ __align__(16) __bf16 pbuf[NWAVES][16 * 32];  // P re-layout 8KB

  const int tid   = threadIdx.x;
  const int lane  = tid & 31;
  const int wid   = tid >> 5;
  const int half  = lane >> 4;
  const int lm    = lane & 15;
  const int b     = blockIdx.x / (SEQ / BLOCK_M);
  const int r0    = (blockIdx.x % (SEQ / BLOCK_M)) * BLOCK_M + wid * 16;
  const int kbase = half * 8;

  const __bf16* Kb = K  + (size_t)b * SEQ * D_HEAD;
  const __bf16* Vb = Vt + (size_t)b * D_HEAD * SEQ;
  const int*    mb = mask + (size_t)b * SEQ;
  __bf16* pw = pbuf[wid];

  // staging geometry: one 16B chunk per thread per tile for each of K and V
  const int ck_key = tid >> 3, ck_dh = (tid & 7) * 8;   // kbuf row 128B
  const int cv_dh  = tid >> 2, cv_k  = (tid & 3) * 8;   // vbuf row  64B
  auto stage = [&](int buf, int kt) {
    async_copy16(&kbuf[buf][ck_key * D_HEAD + ck_dh],
                 Kb + (size_t)(kt + ck_key) * D_HEAD + ck_dh);
    async_copy16(&vbuf[buf][cv_dh * 32 + cv_k],
                 Vb + (size_t)cv_dh * SEQ + kt + cv_k);
  };

  // Q A-fragments for the 2 head-dim k-chunks (resident for whole kernel)
  v16bf qf[2];
  {
    const __bf16* qrow = Q + ((size_t)b * SEQ + (r0 + lm)) * D_HEAD;
#pragma unroll
    for (int c = 0; c < 2; ++c) {
      const __bf16* qp = qrow + c * 32 + kbase;
      qf[c] = combine16(*(const v8bf*)qp, *(const v8bf*)(qp + 16));
    }
  }

  float mrow[8], lrow[8];
#pragma unroll
  for (int v = 0; v < 8; ++v) { mrow[v] = -3.0e38f; lrow[v] = 0.f; }
  v8f o[4];
#pragma unroll
  for (int t = 0; t < 4; ++t) o[t] = v8f{};

  stage(0, 0);

  for (int it = 0; it < SEQ / 32; ++it) {
    const int kt  = it * 32;
    const int buf = it & 1;
    wait_async0();
    __syncthreads();                        // tile(it) visible; tile(it-1) consumed
    if (kt + 32 < SEQ) stage(buf ^ 1, kt + 32);

    // ---- scores: two 16-key subtiles, K-dim = 64 in two chunks of 32 ----
    v8f s[2];
#pragma unroll
    for (int j = 0; j < 2; ++j) {
      s[j] = v8f{};
#pragma unroll
      for (int c = 0; c < 2; ++c) {
        const __bf16* kp = &kbuf[buf][(j * 16 + lm) * D_HEAD + c * 32 + half * 8];
        v16bf kf = combine16(*(const v8bf*)kp, *(const v8bf*)(kp + 16));
        s[j] = __builtin_amdgcn_wmma_f32_16x16x32_bf16(
            false, qf[c], false, kf, (short)0, s[j], false, false);
      }
    }

    // ---- scale + column mask (key = kt + 16j + lm for this lane) ----
    const int mk0 = mb[kt + lm];
    const int mk1 = mb[kt + 16 + lm];
#pragma unroll
    for (int v = 0; v < 8; ++v) {
      s[0][v] = mk0 ? s[0][v] * 0.125f : -1.0e9f;   // 1/sqrt(64)
      s[1][v] = mk1 ? s[1][v] * 0.125f : -1.0e9f;
    }

    // ---- online softmax (per row = v + 8*half, stats replicated per half) ----
    float sc[8];
#pragma unroll
    for (int v = 0; v < 8; ++v) {
      float tm = hmax16(fmaxf(s[0][v], s[1][v]));
      float nm = fmaxf(mrow[v], tm);
      sc[v]    = __expf(mrow[v] - nm);
      float p0 = __expf(s[0][v] - nm);
      float p1 = __expf(s[1][v] - nm);
      s[0][v] = p0; s[1][v] = p1;
      lrow[v] = lrow[v] * sc[v] + hsum16(p0 + p1);
      mrow[v] = nm;
    }
#pragma unroll
    for (int t = 0; t < 4; ++t)
#pragma unroll
      for (int v = 0; v < 8; ++v) o[t][v] *= sc[v];

    // ---- re-layout P: C-layout -> A-layout via wave-private LDS tile ----
#pragma unroll
    for (int j = 0; j < 2; ++j)
#pragma unroll
      for (int v = 0; v < 8; ++v)
        pw[(v + 8 * half) * 32 + j * 16 + lm] = (__bf16)s[j][v];
    asm volatile("s_wait_dscnt 0x0" ::: "memory");  // lanes cross-read the tile
    const __bf16* pr = pw + lm * 32 + kbase;
    v16bf pf = combine16(*(const v8bf*)pr, *(const v8bf*)(pr + 16));

    // ---- O += P * V ----
#pragma unroll
    for (int t = 0; t < 4; ++t) {
      const __bf16* vp = &vbuf[buf][(t * 16 + lm) * 32 + half * 8];
      v16bf vf = combine16(*(const v8bf*)vp, *(const v8bf*)(vp + 16));
      o[t] = __builtin_amdgcn_wmma_f32_16x16x32_bf16(
          false, pf, false, vf, (short)0, o[t], false, false);
    }
  }

  // ---- epilogue: out = O / l, fp32 ----
#pragma unroll
  for (int t = 0; t < 4; ++t)
#pragma unroll
    for (int v = 0; v < 8; ++v) {
      const int r   = r0 + v + 8 * half;
      const int col = t * 16 + lm;
      out[((size_t)b * SEQ + r) * D_HEAD + col] = o[t][v] / lrow[v];
    }
}

// ---------------------------------------------------------------------------
extern "C" void kernel_launch(void* const* d_in, const int* in_sizes, int n_in,
                              void* d_out, int out_size, void* d_ws, size_t ws_size,
                              hipStream_t stream) {
  (void)in_sizes; (void)n_in; (void)out_size; (void)ws_size;
  const float* x    = (const float*)d_in[0];
  const int*   mask = (const int*)d_in[1];
  const float* Wq   = (const float*)d_in[2];
  const float* bq   = (const float*)d_in[3];
  const float* Wk   = (const float*)d_in[4];
  const float* bk   = (const float*)d_in[5];
  const float* Wv   = (const float*)d_in[6];
  const float* bv   = (const float*)d_in[7];
  float* out = (float*)d_out;

  char* ws = (char*)d_ws;
  __bf16* wt  = (__bf16*)(ws + OFF_WT);
  __bf16* Qb  = (__bf16*)(ws + OFF_Q);
  __bf16* Kb  = (__bf16*)(ws + OFF_K);
  __bf16* Vtb = (__bf16*)(ws + OFF_VT);

  wt_convert<<<(3 * WT_ELEMS + 255) / 256, 256, 0, stream>>>(Wq, Wk, Wv, wt);
  qkv_proj<<<BATCH * SEQ / BLOCK_M, 256, 0, stream>>>(x, wt, bq, bk, bv, Qb, Kb, Vtb);
  attention<<<BATCH * SEQ / BLOCK_M, 256, 0, stream>>>(Qb, Kb, Vtb, mask, out);
}
